// GraphMatcher_55542517072535
// MI455X (gfx1250) — matrix operand order repaired and checked by reading, not compiled, not run on hardware
//
#include <hip/hip_runtime.h>
#include <math.h>

// Problem constants (from reference setup_inputs)
#define B_ 4
#define N_ 512
#define K_ 128
#define C_ 256
#define S_ 2

typedef __attribute__((ext_vector_type(16))) _Float16 v16h;
typedef __attribute__((ext_vector_type(8)))  float    v8f;

union V16H { v16h v; _Float16 e[16]; };

// ---------------------------------------------------------------------------
// Kernel A: L2-normalize node features along C.  grid = B*N, block = 256 (=C)
// ---------------------------------------------------------------------------
__global__ void __launch_bounds__(256)
gm_normalize_nodes(const float* __restrict__ nf, float* __restrict__ nnf) {
    const int row = blockIdx.x;              // b*N + n
    const int t   = threadIdx.x;             // c
    float v = nf[(size_t)row * C_ + t];
    __shared__ float red[256];
    red[t] = v * v;
    __syncthreads();
    for (int off = 128; off > 0; off >>= 1) {
        if (t < off) red[t] += red[t + off];
        __syncthreads();
    }
    float d = fmaxf(sqrtf(red[0]), 1e-12f);
    nnf[(size_t)row * C_ + t] = v / d;
}

// ---------------------------------------------------------------------------
// Kernel B: bilinear gather + WMMA dot products.
// grid = B*N*S (blk -> s = blk&1, n = (blk>>1)&511, b = blk>>10), block = 256.
// Wave w handles k-tile w (16 k's). Lane l and l^16 share k-row (l&15) and
// split the C dimension per the 16-bit WMMA A-matrix striping.
//
// Addressing: all lane-divergent terms (corner offsets + lane-half channel
// shift) live in four loop-invariant 32-bit element offsets; the channel
// plane advance is a uniform scalar pointer, so each gather lowers to a
// single global_load_b32 with SGPR base + VGPR offset.
// ---------------------------------------------------------------------------
__global__ void __launch_bounds__(256)
gm_sample_sim(const float* __restrict__ wrel,     // [B,N]
              const float* __restrict__ nnf,      // [B,N,C] (normalized)
              const float* __restrict__ rel8,     // [B,128,128]
              const float* __restrict__ feat8,    // [B,C,128,128]
              const float* __restrict__ rel16,    // [B,64,64]
              const float* __restrict__ feat16,   // [B,C,64,64]
              const float* __restrict__ coords,   // [B,K,N,2]
              float* __restrict__ sims,           // [B,S,N,K]
              float* __restrict__ rels)           // [B,S,N,K]
{
    const int blk = blockIdx.x;
    const int s = blk & 1;
    const int n = (blk >> 1) & (N_ - 1);
    const int b = blk >> 10;                  // N_*S_ = 1024

    // Preload this node's normalized feature row into LDS as f16 (B operand).
    __shared__ _Float16 snnf[C_];
    snnf[threadIdx.x] = (_Float16)nnf[((size_t)b * N_ + n) * C_ + threadIdx.x];
    __syncthreads();

    const int lane = threadIdx.x & 31;
    const int wave = threadIdx.x >> 5;        // k-tile index 0..7
    const int m    = lane & 15;               // k row within tile
    const int hi   = lane >> 4;               // lane half (K-striping select)
    const int k    = wave * 16 + m;

    int H, W;
    const float *relmap, *featmap;
    if (s == 0) { H = 128; W = 128; relmap = rel8;  featmap = feat8;  }
    else        { H = 64;  W = 64;  relmap = rel16; featmap = feat16; }
    const int HW = H * W;

    const float2 g = *(const float2*)(coords + (((size_t)b * K_ + k) * N_ + n) * 2);
    // align_corners=False mapping
    const float ix = ((g.x + 1.0f) * (float)W - 1.0f) * 0.5f;
    const float iy = ((g.y + 1.0f) * (float)H - 1.0f) * 0.5f;
    const float x0f = floorf(ix), y0f = floorf(iy);
    const float x1f = x0f + 1.0f, y1f = y0f + 1.0f;
    const float wx1 = ix - x0f, wx0 = 1.0f - wx1;
    const float wy1 = iy - y0f, wy0 = 1.0f - wy1;

    const float vx0 = (x0f >= 0.0f && x0f <= (float)(W - 1)) ? 1.0f : 0.0f;
    const float vx1 = (x1f >= 0.0f && x1f <= (float)(W - 1)) ? 1.0f : 0.0f;
    const float vy0 = (y0f >= 0.0f && y0f <= (float)(H - 1)) ? 1.0f : 0.0f;
    const float vy1 = (y1f >= 0.0f && y1f <= (float)(H - 1)) ? 1.0f : 0.0f;
    // padding_mode='zeros': fold validity into bilinear weights (no divergence)
    const float w00 = wx0 * wy0 * vx0 * vy0;
    const float w01 = wx1 * wy0 * vx1 * vy0;
    const float w10 = wx0 * wy1 * vx0 * vy1;
    const float w11 = wx1 * wy1 * vx1 * vy1;

    const int x0 = min(max((int)x0f, 0), W - 1);
    const int x1 = min(max((int)x1f, 0), W - 1);
    const int y0 = min(max((int)y0f, 0), H - 1);
    const int y1 = min(max((int)y1f, 0), H - 1);
    const int o00 = y0 * W + x0, o01 = y0 * W + x1;
    const int o10 = y1 * W + x0, o11 = y1 * W + x1;

    // reliability sample (single channel)
    const float* rp = relmap + (size_t)b * HW;
    const float sr = w00 * rp[o00] + w01 * rp[o01] + w10 * rp[o10] + w11 * rp[o11];

    // Loop-invariant per-lane element offsets: corner offset + lane-half
    // channel shift (hi*8 channel planes). 32-bit, max < 16M elements.
    const int lc  = hi * 8 * HW;
    const int i00 = lc + o00, i01 = lc + o01, i10 = lc + o10, i11 = lc + o11;

    const float* fb = featmap + (size_t)b * C_ * HW;   // uniform (scalar)

    v8f  acc   = {};
    float norm2 = 0.0f;

    #pragma unroll
    for (int t = 0; t < 8; ++t) {          // 8 chunks of 32 channels = C_
        const float* bt = fb + (size_t)(t * 32) * HW;  // uniform scalar base
        V16H a, bb;
        // A-matrix 16x32 f16 striping: this lane's channels are
        //   t*32 + hi*8 + j      -> a.e[j]    (j = 0..7)
        //   t*32 + 16 + hi*8 + j -> a.e[8+j]  (j = 0..7)
        #pragma unroll
        for (int j = 0; j < 8; ++j) {
            const float* p0 = bt + j * HW;         // uniform
            float v0 = w00 * p0[i00] + w01 * p0[i01]
                     + w10 * p0[i10] + w11 * p0[i11];
            norm2 += v0 * v0;
            a.e[j] = (_Float16)v0;

            const float* p1 = bt + (16 + j) * HW;  // uniform
            float v1 = w00 * p1[i00] + w01 * p1[i01]
                     + w10 * p1[i10] + w11 * p1[i11];
            norm2 += v1 * v1;
            a.e[8 + j] = (_Float16)v1;
        }
        // B-matrix 32x16 f16 striping matching A: element j holds
        // K = (j>>3)*16 + hi*8 + (j&7).
        // Same data in every column (node feature broadcast).
        #pragma unroll
        for (int j = 0; j < 16; ++j) bb.e[j] = snnf[t * 32 + (j >> 3) * 16 + hi * 8 + (j & 7)];

        acc = __builtin_amdgcn_wmma_f32_16x16x32_f16(
                  false, a.v, false, bb.v, (short)0, acc, false, false);
    }

    // Full squared norm for row m: combine the two lane halves.
    norm2 += __shfl_xor(norm2, 16, 32);

    // Writer lane 0 holds D rows 0..7 in acc[0..7]; lane 16 holds rows 8..15.
    // Fetch the matching norms (row r lives in lane with (lane&15)==r).
    float nrm[8];
    #pragma unroll
    for (int r = 0; r < 8; ++r) nrm[r] = __shfl(norm2, hi * 8 + r, 32);

    const size_t obase = (((size_t)b * S_ + s) * N_ + n) * K_;

    if (hi == 0) {
        rels[obase + k] = wrel[(size_t)b * N_ + n] + sr;
    }
    if (m == 0) {
        const int kb = wave * 16 + hi * 8;
        #pragma unroll
        for (int r = 0; r < 8; ++r) {
            float d = fmaxf(sqrtf(nrm[r]), 1e-12f);
            sims[obase + kb + r] = acc[r] / d;
        }
    }
}

// ---------------------------------------------------------------------------
// Kernel C: per (b,k): softmax over N*S=1024 rels, weighted sum of sims,
// scaled by exp(logit_scale).  grid = B*K, block = 256.
// ---------------------------------------------------------------------------
__global__ void __launch_bounds__(256)
gm_softmax_mix(const float* __restrict__ sims, const float* __restrict__ rels,
               const float* __restrict__ logit_scale, float* __restrict__ logits)
{
    const int bk = blockIdx.x;   // b*K + k
    const int b  = bk >> 7;
    const int k  = bk & (K_ - 1);
    const int t  = threadIdx.x;
    __shared__ float red[256];
    __shared__ float red2[256];

    float mx = -3.4e38f;
    for (int idx = t; idx < N_ * S_; idx += 256)
        mx = fmaxf(mx, rels[((size_t)b * (N_ * S_) + idx) * K_ + k]);
    red[t] = mx; __syncthreads();
    for (int off = 128; off > 0; off >>= 1) {
        if (t < off) red[t] = fmaxf(red[t], red[t + off]);
        __syncthreads();
    }
    const float gmax = red[0];
    __syncthreads();

    float se = 0.0f, sw = 0.0f;
    for (int idx = t; idx < N_ * S_; idx += 256) {
        const size_t off = ((size_t)b * (N_ * S_) + idx) * K_ + k;
        const float p = __expf(rels[off] - gmax);
        se += p;
        sw += p * sims[off];
    }
    red[t] = se; red2[t] = sw; __syncthreads();
    for (int off = 128; off > 0; off >>= 1) {
        if (t < off) { red[t] += red[t + off]; red2[t] += red2[t + off]; }
        __syncthreads();
    }
    if (t == 0) logits[bk] = (red2[0] / red[0]) * __expf(logit_scale[0]);
}

// ---------------------------------------------------------------------------
// Kernel D: final softmax over K per batch.  grid = B, block = 128 (=K)
// ---------------------------------------------------------------------------
__global__ void __launch_bounds__(128)
gm_final_softmax(const float* __restrict__ logits, float* __restrict__ out)
{
    const int b = blockIdx.x;
    const int t = threadIdx.x;
    __shared__ float red[128];
    const float l = logits[b * K_ + t];
    red[t] = l; __syncthreads();
    for (int off = 64; off > 0; off >>= 1) {
        if (t < off) red[t] = fmaxf(red[t], red[t + off]);
        __syncthreads();
    }
    const float gmax = red[0]; __syncthreads();
    const float p = __expf(l - gmax);
    red[t] = p; __syncthreads();
    for (int off = 64; off > 0; off >>= 1) {
        if (t < off) red[t] += red[t + off];
        __syncthreads();
    }
    out[b * K_ + t] = p / red[0];
}

// ---------------------------------------------------------------------------
extern "C" void kernel_launch(void* const* d_in, const int* in_sizes, int n_in,
                              void* d_out, int out_size, void* d_ws, size_t ws_size,
                              hipStream_t stream) {
    const float* wrel   = (const float*)d_in[0];   // [B,N]
    const float* nf     = (const float*)d_in[1];   // [B,N,C]
    const float* rel8   = (const float*)d_in[2];   // [B,128,128]
    const float* feat8  = (const float*)d_in[3];   // [B,C,128,128]
    const float* rel16  = (const float*)d_in[4];   // [B,64,64]
    const float* feat16 = (const float*)d_in[5];   // [B,C,64,64]
    const float* coords = (const float*)d_in[6];   // [B,K,N,2]
    const float* lscale = (const float*)d_in[7];   // scalar
    float* out = (float*)d_out;                    // [B,K]

    // Workspace layout (floats): nnf | sims | rels | logits  (~6.3 MB total)
    float* nnf    = (float*)d_ws;
    float* sims   = nnf  + (size_t)B_ * N_ * C_;
    float* rels   = sims + (size_t)B_ * S_ * N_ * K_;
    float* logits = rels + (size_t)B_ * S_ * N_ * K_;

    gm_normalize_nodes<<<B_ * N_, 256, 0, stream>>>(nf, nnf);
    gm_sample_sim<<<B_ * N_ * S_, 256, 0, stream>>>(wrel, nnf, rel8, feat8,
                                                    rel16, feat16, coords,
                                                    sims, rels);
    gm_softmax_mix<<<B_ * K_, 256, 0, stream>>>(sims, rels, lscale, logits);
    gm_final_softmax<<<B_, 128, 0, stream>>>(logits, out);

    (void)in_sizes; (void)n_in; (void)out_size; (void)ws_size;
}
